// DAAG_Layer_78606491451483
// MI455X (gfx1250) — matrix operand order repaired and checked by reading, not compiled, not run on hardware
//
#include <hip/hip_runtime.h>

typedef __attribute__((ext_vector_type(16))) _Float16 v16h;
typedef __attribute__((ext_vector_type(8)))  _Float16 v8h;
typedef __attribute__((ext_vector_type(8)))  float    v8f;

// D = A(16x32 f16) * B(32x16 f16) + C(16x16 f32)
__device__ __forceinline__ v8f wmma_f16(v16h a, v16h b, v8f c) {
  return __builtin_amdgcn_wmma_f32_16x16x32_f16(false, a, false, b, (short)0, c, false, false);
}

// A fragment: lane L holds row (L&15); halves 0..15 = K kb..kb+15,
// kb = (L>=16)*16.  base is row-major f16 with given stride (in halves).
__device__ __forceinline__ v16h load_a_frag(const _Float16* base, int stride, int lane) {
  const int row = lane & 15;
  const int kb  = (lane >> 4) << 4;
  const v8h lo = *(const v8h*)(base + row * stride + kb);
  const v8h hi = *(const v8h*)(base + row * stride + kb + 8);
  v16h a;
#pragma unroll
  for (int i = 0; i < 8; ++i) { a[i] = lo[i]; a[i + 8] = hi[i]; }
  return a;
}

// B fragment: lane L holds column (L&15), K = kb..kb+15 with kb = (L>=16)*16 + koff.
// baseT is B transposed (row = column of B), contiguous in K, stride in halves.
__device__ __forceinline__ v16h load_b_frag(const _Float16* baseT, int strideT, int lane, int koff) {
  const int col = lane & 15;
  const int kb  = ((lane >> 4) << 4) + koff;
  return *(const v16h*)(baseT + col * strideT + kb);
}

// B=32, SL=64, N=64, IN=64, H=8, FH=32.  One block per (b,sl): grid = 2048, block = 256 (8 waves).
__global__ __launch_bounds__(256)
void daag_layer_kernel(const float* __restrict__ hin,
                       const float* __restrict__ Adj,
                       const float* __restrict__ W_l,
                       const float* __restrict__ W_r,
                       const float* __restrict__ W_v,
                       const float* __restrict__ saw,   // scoreatt_w [64]
                       const float* __restrict__ sab,   // scoreatt_b [64]
                       const float* __restrict__ lnw,   // ln_w [64]
                       const float* __restrict__ lnb,   // ln_b [64]
                       const float* __restrict__ att_w, // [96][32]
                       const float* __restrict__ att_b, // [32]
                       const float* __restrict__ ff1_w, // [32][32]
                       const float* __restrict__ ff1_b, // [32]
                       const float* __restrict__ ff2_w, // [32][32]
                       const float* __restrict__ ff2_b, // [32]
                       const float* __restrict__ rezero,// [32]
                       float* __restrict__ out) {
  __shared__ __attribute__((aligned(32))) _Float16 sh_h[64 * 64];      // h tile f16
  __shared__ __attribute__((aligned(32))) _Float16 sh_w[3][32 * 64];   // W_{l,r,v}[:,head] transposed [n][k]
  __shared__ __attribute__((aligned(32))) _Float16 sh_q[64 * 32];
  __shared__ __attribute__((aligned(32))) _Float16 sh_k[64 * 32];
  __shared__ __attribute__((aligned(32))) _Float16 sh_v[64 * 32];
  __shared__ __attribute__((aligned(32))) _Float16 sh_sm[64 * 64];     // softmax rows (per head)
  __shared__ __attribute__((aligned(32))) _Float16 sh_x[64 * 32];      // att output (f16)
  __shared__ __attribute__((aligned(32))) _Float16 sh_y[64 * 32];      // leaky_relu(ff1)
  __shared__ __attribute__((aligned(32))) _Float16 sh_awT[32 * 96];    // att_w transposed [n][j]
  __shared__ __attribute__((aligned(32))) _Float16 sh_f1T[32 * 32];
  __shared__ __attribute__((aligned(32))) _Float16 sh_f2T[32 * 32];
  __shared__ float sh_se[64];                                          // per-node exp-sum (per head)
  __shared__ float sh_vec[64 * 4 + 32 * 4]; // saw|sab|lnw|lnb|att_b|ff1_b|ff2_b|rezero

  const int tid  = threadIdx.x;
  const int wave = tid >> 5;
  const int lane = tid & 31;
  const int bs   = blockIdx.x;

  const float* hp   = hin + (size_t)bs * 4096;
  const float* adjp = Adj + (size_t)bs * 4096;
  float*       op   = out + (size_t)bs * 16384;

  // L2 warm hints for the (tiny, shared-by-all-blocks) weights
  if (tid < 64) {
    __builtin_prefetch(W_l + tid * 256, 0, 1);
    __builtin_prefetch(W_r + tid * 256, 0, 1);
    __builtin_prefetch(W_v + tid * 256, 0, 1);
  }

  // ---- Phase 0: stage h (f16) + small weights (transposed f16) + vectors ----
  for (int e = tid; e < 4096; e += 256) sh_h[e] = (_Float16)hp[e];
  for (int e = tid; e < 96 * 32; e += 256) { int j = e >> 5, n = e & 31; sh_awT[n * 96 + j] = (_Float16)att_w[j * 32 + n]; }
  for (int e = tid; e < 1024; e += 256) { int k = e >> 5, n = e & 31;
    sh_f1T[n * 32 + k] = (_Float16)ff1_w[e];
    sh_f2T[n * 32 + k] = (_Float16)ff2_w[e];
  }
  if (tid < 64) { sh_vec[tid] = saw[tid]; sh_vec[64 + tid] = sab[tid];
                  sh_vec[128 + tid] = lnw[tid]; sh_vec[192 + tid] = lnb[tid]; }
  if (tid < 32) { sh_vec[256 + tid] = att_b[tid]; sh_vec[288 + tid] = ff1_b[tid];
                  sh_vec[320 + tid] = ff2_b[tid]; sh_vec[352 + tid] = rezero[tid]; }
  __syncthreads();

  const int rt    = wave >> 1;              // row tile 0..3 (16 rows each)
  const int ct    = wave & 1;               // col tile 0..1 (16 cols of FH=32)
  const int colq  = ct * 16 + (lane & 15);  // output column within head
  const int rbase = (lane >> 4) << 3;       // C/D row offset 0 or 8
  const float scale = 0.0078125f;           // 1/sqrt(FH*H*N) = 1/128

  for (int hd = 0; hd < 8; ++hd) {
    // ---- Phase A: stage this head's projection weight columns, transposed ----
    for (int e = tid; e < 3 * 2048; e += 256) {
      int m = e >> 11, r = e & 2047, k = r >> 5, n = r & 31;
      const float* W = (m == 0) ? W_l : (m == 1) ? W_r : W_v;
      sh_w[m][n * 64 + k] = (_Float16)W[k * 256 + hd * 32 + n];
    }
    if (tid < 64) sh_se[tid] = 0.f;
    __syncthreads();

    // ---- Phase B: q_h, k_h(raw), v_h = h @ W  (64x64 @ 64x32) ----
    {
      v16h a0 = load_a_frag(sh_h + rt * 16 * 64 + 0,  64, lane);
      v16h a1 = load_a_frag(sh_h + rt * 16 * 64 + 32, 64, lane);
#pragma unroll
      for (int m = 0; m < 3; ++m) {
        v8f acc = {};
        acc = wmma_f16(a0, load_b_frag(sh_w[m] + ct * 16 * 64, 64, lane, 0),  acc);
        acc = wmma_f16(a1, load_b_frag(sh_w[m] + ct * 16 * 64, 64, lane, 32), acc);
        _Float16* dst = (m == 0) ? sh_q : (m == 1) ? sh_k : sh_v;
#pragma unroll
        for (int r = 0; r < 8; ++r)
          dst[(rt * 16 + rbase + r) * 32 + colq] = (_Float16)acc[r];
      }
    }
    __syncthreads();

    // ---- Phase C: score = q @ k_raw^T; masked exp-sum (note k = -k_raw) ----
#pragma unroll
    for (int tt = 0; tt < 2; ++tt) {
      int t   = wave * 2 + tt;
      int rti = t >> 2;   // node-i tile
      int ctj = t & 3;    // node-j tile
      v16h aq = load_a_frag(sh_q + rti * 16 * 32, 32, lane);
      v16h bk = load_b_frag(sh_k + ctj * 16 * 32, 32, lane, 0); // k row-major == k^T transposed
      v8f acc = {};
      acc = wmma_f16(aq, bk, acc);
      int jj = ctj * 16 + (lane & 15);
#pragma unroll
      for (int r = 0; r < 8; ++r) {
        int ii = rti * 16 + rbase + r;
        float am  = adjp[ii * 64 + jj];
        float val = (ii != jj && am > 0.5f) ? __expf(-scale * acc[r]) : 0.f;
        val += __shfl_xor(val, 8, 32);
        val += __shfl_xor(val, 4, 32);
        val += __shfl_xor(val, 2, 32);
        val += __shfl_xor(val, 1, 32);
        if ((lane & 15) == 0) atomicAdd(&sh_se[ii], val);
      }
    }
    __syncthreads();

    // ---- Phase D: per node row: outer-product + LayerNorm + softmax (len 64) ----
    if (tid < 64) {
      const float se = sh_se[tid];
      float sum = 0.f, sumsq = 0.f;
      for (int j = 0; j < 64; ++j) {
        float t = fmaf(se, sh_vec[j], sh_vec[64 + j]);
        sum += t; sumsq += t * t;
      }
      float mean = sum * 0.015625f;
      float var  = sumsq * 0.015625f - mean * mean;
      float rstd = rsqrtf(var + 1e-5f);
      float mx = -3.4e38f;
      for (int j = 0; j < 64; ++j) {
        float t = fmaf(se, sh_vec[j], sh_vec[64 + j]);
        float y = (t - mean) * rstd * sh_vec[128 + j] + sh_vec[192 + j];
        mx = fmaxf(mx, y);
      }
      float es = 0.f;
      for (int j = 0; j < 64; ++j) {
        float t = fmaf(se, sh_vec[j], sh_vec[64 + j]);
        float y = (t - mean) * rstd * sh_vec[128 + j] + sh_vec[192 + j];
        float e = __expf(y - mx);
        es += e;
        sh_sm[tid * 64 + j] = (_Float16)e;
      }
      float inv = 1.f / es;
      for (int j = 0; j < 64; ++j)
        sh_sm[tid * 64 + j] = (_Float16)((float)sh_sm[tid * 64 + j] * inv);
    }
    __syncthreads();

    // ---- Phase E: s_att = sm @ att_w[:64] + v @ att_w[64:] + att_b ----
    float satt[8];
    {
      v8f acc = {};
      acc = wmma_f16(load_a_frag(sh_sm + rt * 16 * 64 + 0,  64, lane),
                     load_b_frag(sh_awT + ct * 16 * 96, 96, lane, 0),  acc);
      acc = wmma_f16(load_a_frag(sh_sm + rt * 16 * 64 + 32, 64, lane),
                     load_b_frag(sh_awT + ct * 16 * 96, 96, lane, 32), acc);
      acc = wmma_f16(load_a_frag(sh_v + rt * 16 * 32, 32, lane),
                     load_b_frag(sh_awT + ct * 16 * 96, 96, lane, 64), acc);
      float bb = sh_vec[256 + colq];
#pragma unroll
      for (int r = 0; r < 8; ++r) {
        satt[r] = acc[r] + bb;
        sh_x[(rt * 16 + rbase + r) * 32 + colq] = (_Float16)satt[r];
      }
    }
    __syncthreads();

    // ---- Phase F: ff1 + leaky relu ----
    {
      v8f acc = {};
      acc = wmma_f16(load_a_frag(sh_x + rt * 16 * 32, 32, lane),
                     load_b_frag(sh_f1T + ct * 16 * 32, 32, lane, 0), acc);
      float bb = sh_vec[288 + colq];
#pragma unroll
      for (int r = 0; r < 8; ++r) {
        float t = acc[r] + bb;
        t = (t > 0.f) ? t : 0.01f * t;
        sh_y[(rt * 16 + rbase + r) * 32 + colq] = (_Float16)t;
      }
    }
    __syncthreads();

    // ---- Phase G: ff2 + rezero residual + store ----
    {
      v8f acc = {};
      acc = wmma_f16(load_a_frag(sh_y + rt * 16 * 32, 32, lane),
                     load_b_frag(sh_f2T + ct * 16 * 32, 32, lane, 0), acc);
      float b2 = sh_vec[320 + colq];
      float rz = sh_vec[352 + colq];
#pragma unroll
      for (int r = 0; r < 8; ++r) {
        int row = rt * 16 + rbase + r;
        op[row * 256 + hd * 32 + colq] = satt[r] + rz * (acc[r] + b2);
      }
    }
    __syncthreads();
  }
}

extern "C" void kernel_launch(void* const* d_in, const int* in_sizes, int n_in,
                              void* d_out, int out_size, void* d_ws, size_t ws_size,
                              hipStream_t stream) {
  (void)in_sizes; (void)n_in; (void)out_size; (void)d_ws; (void)ws_size;
  const float* hin    = (const float*)d_in[0];
  const float* Adj    = (const float*)d_in[1];
  const float* W_l    = (const float*)d_in[2];
  const float* W_r    = (const float*)d_in[3];
  const float* W_v    = (const float*)d_in[4];
  const float* saw    = (const float*)d_in[5];
  const float* sab    = (const float*)d_in[6];
  const float* lnw    = (const float*)d_in[7];
  const float* lnb    = (const float*)d_in[8];
  const float* att_w  = (const float*)d_in[9];
  const float* att_b  = (const float*)d_in[10];
  const float* ff1_w  = (const float*)d_in[11];
  const float* ff1_b  = (const float*)d_in[12];
  const float* ff2_w  = (const float*)d_in[13];
  const float* ff2_b  = (const float*)d_in[14];
  const float* rezero = (const float*)d_in[15];
  float* outp = (float*)d_out;

  daag_layer_kernel<<<dim3(32 * 64), dim3(256), 0, stream>>>(
      hin, Adj, W_l, W_r, W_v, saw, sab, lnw, lnb,
      att_w, att_b, ff1_w, ff1_b, ff2_w, ff2_b, rezero, outp);
}
